// ModelWrapper_37074157699080
// MI455X (gfx1250) — matrix operand order repaired and checked by reading, not compiled, not run on hardware
//
#include <hip/hip_runtime.h>
#include <hip/hip_bf16.h>

typedef _Float16 v16h  __attribute__((ext_vector_type(16)));
typedef _Float16 half8 __attribute__((ext_vector_type(8)));
typedef float    v8f   __attribute__((ext_vector_type(8)));

#define NANCH 5376
#define NPAD  8192
#define MAXDET 300
#define NDET  304          // 300 padded to 16-multiple for WMMA N tiles
#define NTILES_N 19        // 304/16
#define HW 16384           // 128*128 proto pixels
#define OUT_MASK_OFF 1800  // 300 + 300 + 1200

// ---------------------------------------------------------------- prep ----
__global__ void prep_kernel(const float* __restrict__ boxes_in,
                            float* __restrict__ sc, int* __restrict__ cls,
                            float* __restrict__ bx)
{
    int n = blockIdx.x * blockDim.x + threadIdx.x;
    if (n >= NANCH) return;
    float cx = boxes_in[0 * NANCH + n];
    float cy = boxes_in[1 * NANCH + n];
    float w  = boxes_in[2 * NANCH + n];
    float h  = boxes_in[3 * NANCH + n];
    float best = boxes_in[4 * NANCH + n];
    int bi = 0;
    #pragma unroll
    for (int c = 1; c < 4; ++c) {
        float p = boxes_in[(4 + c) * NANCH + n];
        if (p > best) { best = p; bi = c; }   // strict > == first-occurrence argmax
    }
    sc[n]  = best;
    cls[n] = bi;
    float x1 = cx - 0.5f * w, y1 = cy - 0.5f * h;
    float x2 = cx + 0.5f * w, y2 = cy + 0.5f * h;
    // store in b_nms order: (y1, x1, y2, x2)
    bx[n * 4 + 0] = y1; bx[n * 4 + 1] = x1;
    bx[n * 4 + 2] = y2; bx[n * 4 + 3] = x2;
}

// ------------------------------------------------------ bitonic sort ------
__global__ __launch_bounds__(1024)
void sort_kernel(const float* __restrict__ sc, int* __restrict__ sorted_idx)
{
    __shared__ unsigned long long keys[NPAD];   // 64 KB LDS
    const int t = threadIdx.x;
    for (int i = t; i < NPAD; i += 1024) {
        unsigned long long k = 0ull;
        if (i < NANCH) {
            unsigned int sb = __float_as_uint(sc[i]);  // scores >= 0 -> monotone bits
            k = ((unsigned long long)sb << 32) |
                (unsigned long long)(0xFFFFFFFFu - (unsigned)i);
        }
        keys[i] = k;
    }
    __syncthreads();
    for (int k = 2; k <= NPAD; k <<= 1) {
        for (int j = k >> 1; j > 0; j >>= 1) {
            for (int i = t; i < NPAD; i += 1024) {
                int ixj = i ^ j;
                if (ixj > i) {
                    bool descBlock = ((i & k) == 0);
                    unsigned long long a = keys[i], b = keys[ixj];
                    if ((a < b) == descBlock) { keys[i] = b; keys[ixj] = a; }
                }
            }
            __syncthreads();
        }
    }
    for (int i = t; i < NANCH; i += 1024)
        sorted_idx[i] = (int)(0xFFFFFFFFu -
                              (unsigned)(keys[i] & 0xFFFFFFFFull));
}

// ------------------------------------------------------ greedy NMS --------
__global__ __launch_bounds__(1024)
void nms_kernel(const float* __restrict__ sc, const float* __restrict__ bx,
                const int* __restrict__ sorted_idx,
                int* __restrict__ sel, int* __restrict__ sel_count)
{
    __shared__ float pb[4];
    __shared__ int s_i, s_cnt, s_cur;
    __shared__ unsigned char alive[NANCH];
    const int t = threadIdx.x;

    float y1[6], x1[6], y2[6], x2[6], ar[6];
    #pragma unroll
    for (int k = 0; k < 6; ++k) {
        int j = t + k * 1024;
        if (j < NANCH) {
            int oi = sorted_idx[j];
            float a0 = bx[oi * 4 + 0], a1 = bx[oi * 4 + 1];
            float a2 = bx[oi * 4 + 2], a3 = bx[oi * 4 + 3];
            y1[k] = a0; x1[k] = a1; y2[k] = a2; x2[k] = a3;
            ar[k] = (a2 - a0) * (a3 - a1);
            alive[j] = (sc[oi] >= 0.25f) ? 1 : 0;   // score threshold
        }
    }
    if (t == 0) { s_cur = 0; s_cnt = 0; }
    __syncthreads();

    for (;;) {
        if (t == 0) {
            int i = s_cur;
            while (i < NANCH && !alive[i]) ++i;
            if (i >= NANCH || s_cnt >= MAXDET) {
                s_i = -1;
            } else {
                s_i = i;
                int oi = sorted_idx[i];
                sel[s_cnt] = oi;
                s_cnt = s_cnt + 1;
                pb[0] = bx[oi * 4 + 0]; pb[1] = bx[oi * 4 + 1];
                pb[2] = bx[oi * 4 + 2]; pb[3] = bx[oi * 4 + 3];
                s_cur = i + 1;
            }
        }
        __syncthreads();
        const int i = s_i;
        if (i < 0) break;
        const float py1 = pb[0], px1 = pb[1], py2 = pb[2], px2 = pb[3];
        const float pa = (py2 - py1) * (px2 - px1);
        #pragma unroll
        for (int k = 0; k < 6; ++k) {
            int j = t + k * 1024;
            if (j < NANCH && j > i && alive[j]) {
                float iy1 = fmaxf(py1, y1[k]);
                float ix1 = fmaxf(px1, x1[k]);
                float iy2 = fminf(py2, y2[k]);
                float ix2 = fminf(px2, x2[k]);
                float inter = fmaxf(iy2 - iy1, 0.0f) * fmaxf(ix2 - ix1, 0.0f);
                float iou = inter / (pa + ar[k] - inter + 1e-9f);
                if (iou > 0.75f) alive[j] = 0;
            }
        }
        __syncthreads();
    }
    if (t == 0) *sel_count = s_cnt;
}

// ------------------------------------------------------ gather ------------
__global__ void gather_kernel(const float* __restrict__ boxes_in,
                              const float* __restrict__ sc,
                              const int* __restrict__ cls,
                              const float* __restrict__ bx,
                              const int* __restrict__ sel,
                              const int* __restrict__ sel_count,
                              float* __restrict__ vf, float* __restrict__ bb,
                              _Float16* __restrict__ coeffh,
                              float* __restrict__ out)
{
    int s = blockIdx.x * blockDim.x + threadIdx.x;
    if (s >= NDET) return;
    int cnt = *sel_count;
    bool valid = (s < cnt) && (s < MAXDET);
    float vfv = valid ? 1.0f : 0.0f;
    int oi = valid ? sel[s] : 0;
    vf[s] = vfv;
    #pragma unroll
    for (int c = 0; c < 4; ++c) {
        float b = bx[oi * 4 + c] * vfv;
        bb[s * 4 + c] = b * 0.25f;                 // ratio = 512/128 = 4
        if (s < MAXDET) out[600 + s * 4 + c] = b;  // boxes_sel
    }
    if (s < MAXDET) {
        out[s] = sc[oi] * vfv;                     // scores_sel
        out[300 + s] = (float)cls[oi] * vfv;       // cls_sel
    }
    for (int k = 0; k < 32; ++k) {
        float cv = valid ? boxes_in[(8 + k) * NANCH + oi] : 0.0f;
        coeffh[s * 32 + k] = (_Float16)cv;         // column-major [n][k]
    }
}

// ------------------------------------------------ mask GEMM (WMMA) --------
// sigmoid(proto[16384x32] @ coeff[32x304]) -> threshold/crop -> sum over k.
// One wave per 16-pixel M tile; A fragment converted once, 19 WMMAs over all
// N tiles, crop-sum accumulated in registers; one butterfly + direct store.
__global__ __launch_bounds__(256)
void masks_wmma_kernel(const float* __restrict__ proto,
                       const _Float16* __restrict__ coeffh,
                       const float* __restrict__ vf,
                       const float* __restrict__ bb,
                       float* __restrict__ masks)
{
    const int lane = threadIdx.x & 31;
    const int wid  = blockIdx.x * (blockDim.x >> 5) + (threadIdx.x >> 5); // 0..1023
    const int half = lane >> 4;
    const int nloc = lane & 15;

    const int tbase = wid * 16;               // 16 consecutive pixels, same row
    const float rowf  = (float)(tbase >> 7);  // constant mask row for this tile
    const float cbase = (float)(tbase & 127); // first mask column of the tile

    // A fragment: row m = lane&15, K in [8h,8h+8) then [16+8h,16+8h+8)
    const float* prow = proto + (tbase + (lane & 15)) * 32;
    union { v16h v; _Float16 e[16]; } Af;
    #pragma unroll
    for (int q = 0; q < 8; ++q) {
        Af.e[q]     = (_Float16)prow[8 * half + q];
        Af.e[8 + q] = (_Float16)prow[16 + 8 * half + q];
    }

    float acc[8];
    #pragma unroll
    for (int i = 0; i < 8; ++i) acc[i] = 0.0f;

    for (int nt = 0; nt < NTILES_N; ++nt) {
        const int n = nt * 16 + nloc;
        // B fragment: lane holds K in [8h,8h+8) then [16+8h,16+8h+8) of column n
        union { v16h v; half8 h[2]; } Bf;
        const _Float16* crow = coeffh + n * 32;
        Bf.h[0] = *(const half8*)(crow + 8 * half);
        Bf.h[1] = *(const half8*)(crow + 16 + 8 * half);

        const float vfn = vf[n];
        const float b0 = bb[n * 4 + 0], b1 = bb[n * 4 + 1];
        const float b2 = bb[n * 4 + 2], b3 = bb[n * 4 + 3];
        // row-crop is uniform across the tile: fold validity into it
        const float rowfac = ((b0 <= rowf) && (rowf <= b2)) ? vfn : 0.0f;

        v8f c = {};
        c = __builtin_amdgcn_wmma_f32_16x16x32_f16(
                false, Af.v, false, Bf.v, (short)0, c, false, false);

        #pragma unroll
        for (int i = 0; i < 8; ++i) {
            const float x = c[i];
            const float cf = cbase + (float)(i + 8 * half); // mask column
            // sigmoid(x) >= 0.5  <=>  x >= 0 ; v_rcp instead of IEEE divide
            float s = __builtin_amdgcn_rcpf(1.0f + __expf(-x));
            float val = (x >= 0.0f) ? s : 0.0f;
            val *= rowfac;
            val = ((b1 <= cf) && (cf <= b3)) ? val : 0.0f;
            acc[i] += val;
        }
    }

    // one butterfly over the 16 lanes sharing each pixel, then direct store
    #pragma unroll
    for (int i = 0; i < 8; ++i) {
        float v = acc[i];
        #pragma unroll
        for (int off = 1; off < 16; off <<= 1)
            v += __shfl_xor(v, off, 32);
        if (nloc == 0)
            masks[tbase + i + 8 * half] = v;   // every pixel written exactly once
    }
}

// ------------------------------------------------------ launcher ----------
extern "C" void kernel_launch(void* const* d_in, const int* in_sizes, int n_in,
                              void* d_out, int out_size, void* d_ws, size_t ws_size,
                              hipStream_t stream)
{
    const float* boxes_in = (const float*)d_in[0];   // (1,40,5376)
    const float* proto    = (const float*)d_in[1];   // (1,128,128,32)
    float* out = (float*)d_out;                      // 300+300+1200+16384

    char* ws = (char*)d_ws;
    float*    sc         = (float*)(ws + 0);         // 5376 f32
    int*      cls        = (int*)(ws + 21504);       // 5376 i32
    float*    bx         = (float*)(ws + 43008);     // 5376*4 f32
    int*      sorted_idx = (int*)(ws + 129024);      // 5376 i32
    int*      sel        = (int*)(ws + 150528);      // 300 i32
    int*      sel_count  = (int*)(ws + 151808);      // 1 i32
    float*    vf         = (float*)(ws + 151936);    // 304 f32
    float*    bb         = (float*)(ws + 153216);    // 304*4 f32
    _Float16* coeffh     = (_Float16*)(ws + 158208); // 304*32 f16

    prep_kernel<<<NANCH / 256, 256, 0, stream>>>(boxes_in, sc, cls, bx);
    sort_kernel<<<1, 1024, 0, stream>>>(sc, sorted_idx);
    nms_kernel<<<1, 1024, 0, stream>>>(sc, bx, sorted_idx, sel, sel_count);
    gather_kernel<<<2, 256, 0, stream>>>(boxes_in, sc, cls, bx, sel, sel_count,
                                         vf, bb, coeffh, out);
    // 1024 waves, one 16-pixel tile each: 128 blocks x 8 waves
    masks_wmma_kernel<<<128, 256, 0, stream>>>(proto, coeffh, vf, bb,
                                               out + OUT_MASK_OFF);
}
